// SelfAttention_12206297055710
// MI455X (gfx1250) — compile-verified, matrix operand order checked by reading, not run on hardware
//
#include <hip/hip_runtime.h>
#include <hip/hip_bf16.h>
#include <math.h>

// ---------------------------------------------------------------------------
// Adaptive-span attention for MI455X (gfx1250, wave32, WMMA).
//   logits[m][p] = q[m] . (key[b,p] + key_pe[p]) / sqrt(H)   for p in [lo,hi]
//   softmax additionally includes (max_span - Nvalid) zero-logit slots.
// Flash-style online softmax over 32-position chunks, bf16 WMMA for both
// Q.K^T and P.V, f32 accumulation.
// ---------------------------------------------------------------------------

typedef __bf16 bf16_t;
typedef __attribute__((ext_vector_type(8)))  __bf16 bf16x8;
typedef __attribute__((ext_vector_type(16))) __bf16 v16bf;
typedef __attribute__((ext_vector_type(8)))  float  v8f;

constexpr int B = 2, M = 2048, H = 128;
constexpr int TILES = B * (M / 16);          // 256 16-token tiles
constexpr int WPB   = 4;                     // waves per block

// Load a v16bf WMMA operand: two contiguous 8-element (16B) runs 16 elements
// apart -- matches the 16-bit A/B VGPR layout (K pairs split across lane halves).
__device__ inline v16bf ld_pair16(const bf16_t* p) {
  bf16x8 lo = *reinterpret_cast<const bf16x8*>(p);
  bf16x8 hi = *reinterpret_cast<const bf16x8*>(p + 16);
  v16bf r;
#pragma unroll
  for (int e = 0; e < 8; ++e) { r[e] = lo[e]; r[e + 8] = hi[e]; }
  return r;
}

__global__ void k_span(const float* __restrict__ span, int* __restrict__ ms) {
  int t = blockIdx.x * blockDim.x + threadIdx.x;
  if (t >= B * M) return;
  float s0 = span[2 * t], s1 = span[2 * t + 1];
  int left  = (int)rintf(fmaxf(s0, -s1));
  int right = (int)rintf(fmaxf(-s0, s1));
  int len = left + right; len = (len < 0 ? -len : len) + 1;
  atomicMax(ms, len);
}

__global__ void k_convert(const float* __restrict__ q, const float* __restrict__ k,
                          const float* __restrict__ v, const float* __restrict__ pe,
                          bf16_t* __restrict__ Qb, bf16_t* __restrict__ Kb,
                          bf16_t* __restrict__ Vt) {
  int i = blockIdx.x * blockDim.x + threadIdx.x;   // over B*M*H
  if (i >= B * M * H) return;
  int h  = i & (H - 1);
  int bm = i >> 7;            // /H
  int mm = bm & (M - 1);
  int bb = bm >> 11;          // /M
  Qb[i] = (bf16_t)q[i];
  Kb[i] = (bf16_t)(k[i] + pe[mm * H + h]);                 // Keff = key + pe
  Vt[((size_t)(bb * H + h)) * M + mm] = (bf16_t)v[i];      // transposed V
}

__global__ void __launch_bounds__(32 * WPB)
k_attn(const bf16_t* __restrict__ Qb, const bf16_t* __restrict__ Kb,
       const bf16_t* __restrict__ Vt, const float* __restrict__ span,
       const int* __restrict__ msp, float* __restrict__ out) {
  __shared__ __align__(16) bf16_t Plds[WPB][16][32];   // P transpose bounce
  const int wave = threadIdx.x >> 5;
  const int lane = threadIdx.x & 31;
  const int tile = blockIdx.x * WPB + wave;
  if (tile >= TILES) return;                 // uniform per wave
  const int b    = tile / (M / 16);
  const int m0   = (tile % (M / 16)) * 16;
  const int half = (lane >= 16) ? 1 : 0;     // C-layout lane half
  const int col  = lane & 15;                // C-layout column / A,B row-col
  const int koff = half * 8;                 // K offset for upper lane half

  int   lo[8], hi[8];
  float mrun[8], drun[8];
  int lmin = M - 1, lmax = 0;
#pragma unroll
  for (int v = 0; v < 8; ++v) {
    int tok = m0 + v + 8 * half;             // row this (vgpr,lanehalf) owns
    float s0 = span[(b * M + tok) * 2 + 0];
    float s1 = span[(b * M + tok) * 2 + 1];
    int left  = (int)rintf(fmaxf(s0, -s1));
    int right = (int)rintf(fmaxf(-s0, s1));
    int len = left + right; len = (len < 0 ? -len : len) + 1;
    int first = tok - left;
    int l = max(0, first);
    int h = min(M - 1, first + len - 1);
    if (h < l) h = l;
    lo[v] = l; hi[v] = h;
    mrun[v] = -1e30f; drun[v] = 0.0f;
    lmin = min(lmin, l); lmax = max(lmax, h);
  }
#pragma unroll
  for (int mk = 1; mk < 32; mk <<= 1) {      // wave-wide union window
    lmin = min(lmin, __shfl_xor(lmin, mk));
    lmax = max(lmax, __shfl_xor(lmax, mk));
  }

  // Q A-operands (16 tokens x 128), resident for the whole loop.
  v16bf qa[4];
#pragma unroll
  for (int c = 0; c < 4; ++c)
    qa[c] = ld_pair16(Qb + ((size_t)(b * M + m0 + col)) * H + c * 32 + koff);

  v8f acc[8] = {};                           // 16 tokens x 128 f32 output acc
  const float scale = 0.08838834764831845f;  // 1/sqrt(128)

  for (int p0 = (lmin & ~31); p0 <= lmax; p0 += 32) {
    // ---- logits: S(16x32) = Q(16x128) . Keff^T, two 16x16 N-tiles ----
    v8f sA = {}, sB = {};
    const bf16_t* kA = Kb + ((size_t)(b * M + min(p0 + col,      M - 1))) * H + koff;
    const bf16_t* kB = Kb + ((size_t)(b * M + min(p0 + 16 + col, M - 1))) * H + koff;
#pragma unroll
    for (int c = 0; c < 4; ++c) {
      v16bf b0 = ld_pair16(kA + c * 32);
      v16bf b1 = ld_pair16(kB + c * 32);
      sA = __builtin_amdgcn_wmma_f32_16x16x32_bf16(false, qa[c], false, b0, (short)0, sA, false, false);
      sB = __builtin_amdgcn_wmma_f32_16x16x32_bf16(false, qa[c], false, b1, (short)0, sB, false, false);
    }
    __builtin_prefetch(Kb + ((size_t)(b * M + min(p0 + 32 + col, M - 1))) * H, 0, 1);

    // ---- masked online softmax (per-row state replicated across 16 lanes) ----
    float rescale[8];
#pragma unroll
    for (int v = 0; v < 8; ++v) {
      int  pA = p0 + col, pB = p0 + 16 + col;
      bool va = (pA >= lo[v]) && (pA <= hi[v]);
      bool vb = (pB >= lo[v]) && (pB <= hi[v]);
      float sa = va ? sA[v] * scale : -1e30f;
      float sb = vb ? sB[v] * scale : -1e30f;
      float cm = fmaxf(sa, sb);
#pragma unroll
      for (int mk = 1; mk < 16; mk <<= 1) cm = fmaxf(cm, __shfl_xor(cm, mk));
      float nm = fmaxf(mrun[v], cm);
      float r  = __expf(mrun[v] - nm);
      mrun[v] = nm; rescale[v] = r;
      float pa = va ? __expf(sa - nm) : 0.0f;
      float pb = vb ? __expf(sb - nm) : 0.0f;
      float rs = pa + pb;
#pragma unroll
      for (int mk = 1; mk < 16; mk <<= 1) rs += __shfl_xor(rs, mk);
      drun[v] = drun[v] * r + rs;
      int row = v + 8 * half;                // C-layout -> [row][chunk-col]
      Plds[wave][row][col]      = (bf16_t)pa;
      Plds[wave][row][col + 16] = (bf16_t)pb;
    }
#pragma unroll
    for (int t = 0; t < 8; ++t)
#pragma unroll
      for (int v = 0; v < 8; ++v) acc[t][v] *= rescale[v];

    // ---- O += P(16x32) . V(32x128); P re-read from LDS in A-layout ----
    // (DS ops from one wave complete in order: stores above precede this load)
    v16bf pA16 = ld_pair16(&Plds[wave][col][koff]);
#pragma unroll
    for (int t = 0; t < 8; ++t) {
      const bf16_t* vp = Vt + ((size_t)(b * H + 16 * t + col)) * M + p0 + koff;
      v16bf vB = ld_pair16(vp);
      acc[t] = __builtin_amdgcn_wmma_f32_16x16x32_bf16(false, pA16, false, vB, (short)0, acc[t], false, false);
    }
  }

  // ---- epilogue: fold in the (max_span - Nvalid) zero-logit slots ----
  const int mspan = *msp;
  float rowfac[8];
#pragma unroll
  for (int v = 0; v < 8; ++v) {
    int nz = mspan - (hi[v] - lo[v] + 1);
    float m  = mrun[v];
    float nm = (nz > 0) ? fmaxf(m, 0.0f) : m;
    float r  = __expf(m - nm);
    float d  = drun[v] * r + (float)nz * __expf(-nm);
    rowfac[v] = r / d;
  }
#pragma unroll
  for (int t = 0; t < 8; ++t)
#pragma unroll
    for (int v = 0; v < 8; ++v) {
      int tok = m0 + v + 8 * half;
      out[((size_t)(b * M + tok)) * H + 16 * t + col] = acc[t][v] * rowfac[v];
    }
}

extern "C" void kernel_launch(void* const* d_in, const int* in_sizes, int n_in,
                              void* d_out, int out_size, void* d_ws, size_t ws_size,
                              hipStream_t stream) {
  (void)in_sizes; (void)n_in; (void)out_size; (void)ws_size;
  const float* q    = (const float*)d_in[0];   // [B,M,H]
  const float* k    = (const float*)d_in[1];   // [B,M,H]
  const float* v    = (const float*)d_in[2];   // [B,M,H]
  const float* span = (const float*)d_in[3];   // [B,M,2]
  const float* pe   = (const float*)d_in[4];   // [1,M,H]
  float* out = (float*)d_out;                  // [B,M,H] f32

  // workspace layout: [0,256) max_span int; then bf16 Qb, Kb(=key+pe), Vt(+pad)
  char* ws = (char*)d_ws;
  int*    msp = (int*)ws;
  bf16_t* Qb  = (bf16_t*)(ws + 256);
  bf16_t* Kb  = Qb + (size_t)B * M * H;
  bf16_t* Vt  = Kb + (size_t)B * M * H;

  hipMemsetAsync(d_ws, 0, 256, stream);
  k_span   <<<(B * M + 255) / 256,     256, 0, stream>>>(span, msp);
  k_convert<<<(B * M * H + 255) / 256, 256, 0, stream>>>(q, k, v, pe, Qb, Kb, Vt);
  k_attn   <<<TILES / WPB, 32 * WPB, 0, stream>>>(Qb, Kb, Vt, span, msp, out);
}
